// ADSR_78546361909565
// MI455X (gfx1250) — compile-verified, matrix operand order checked by reading, not run on hardware
//
#include <hip/hip_runtime.h>
#include <math.h>

// ---------------------------------------------------------------------------
// ADSR envelope, B=64 voices x T=131072 samples.
// Pass 1: row-sum of gate (== note-off index per row) via WMMA reduction,
//         staged through LDS with gfx1250 async global->LDS loads.
// Pass 2: closed-form envelope, write-only 32MB (never re-reads gate).
// Total HBM traffic ~64MB -> ~2.9us roofline at 23.3 TB/s.
// ---------------------------------------------------------------------------

typedef __attribute__((ext_vector_type(16))) _Float16 v16h;
typedef __attribute__((ext_vector_type(8)))  float    v8f;
typedef __attribute__((ext_vector_type(4)))  int      v4i;

#ifndef __has_builtin
#define __has_builtin(x) 0
#endif

#if __has_builtin(__builtin_amdgcn_global_load_async_to_lds_b128)
#define USE_ASYNC_LDS 1
#else
#define USE_ASYNC_LDS 0
#endif

// Pointer types matching the builtin's parameters:
//   param 0: int4 in global (__device__ / addrspace(1))
//   param 1: int4 in LDS (addrspace(3))
typedef __attribute__((address_space(1))) v4i gv4i_t;
typedef __attribute__((address_space(3))) v4i sv4i_t;

__device__ __forceinline__ void wait_async_zero() {
#if __has_builtin(__builtin_amdgcn_s_wait_asynccnt)
  __builtin_amdgcn_s_wait_asynccnt(0);
#else
  asm volatile("s_wait_asynccnt 0" ::: "memory");
#endif
}

#define NB 64
#define NT 131072
#define ROWS 16            // rows per row-group (WMMA M)
#define TILE_COLS 512      // columns staged per block per step
#define LDS_STRIDE 516     // +4 floats pad -> conflict-free ds_load_b128 gather
#define STEPS 4
#define COLS_PER_BLOCK (TILE_COLS * STEPS)   // 2048 cols per block

// --------------------------- pass 0: zero workspace -------------------------
__global__ void zero_rowsum_kernel(float* __restrict__ rs) {
  if (threadIdx.x < NB) rs[threadIdx.x] = 0.0f;
}

// --------------------------- pass 1: WMMA row-sum ---------------------------
// grid = (NT/COLS_PER_BLOCK = 64, NB/ROWS = 4), block = 256 (8 wave32)
__global__ __launch_bounds__(256) void rowsum_wmma_kernel(
    const float* __restrict__ gate, float* __restrict__ rowsum)
{
  __shared__ alignas(16) float tile[ROWS * LDS_STRIDE];

  const int tid  = threadIdx.x;
  const int lane = tid & 31;
  const int wave = tid >> 5;
  const int m    = lane & 15;   // WMMA A-layout: row within 16-row group
  const int h    = lane >> 4;   // WMMA A-layout: K-half select
  const size_t row0 = (size_t)blockIdx.y * ROWS;
  const size_t col0 = (size_t)blockIdx.x * COLS_PER_BLOCK;

  v16h ones;
#pragma unroll
  for (int i = 0; i < 16; ++i) ones[i] = (_Float16)1.0f;
  v8f acc = {};

  for (int step = 0; step < STEPS; ++step) {
    const size_t cbase = col0 + (size_t)step * TILE_COLS;

    // Stage 16x512 f32 tile: 2048 16B-chunks, 8 per thread, coalesced.
#pragma unroll
    for (int j = 0; j < 8; ++j) {
      const int k  = tid + 256 * j;
      const int r  = k >> 7;          // row 0..15 (128 chunks per row)
      const int cc = (k & 127) * 4;   // float column within row
      const float* src = gate + (row0 + (size_t)r) * (size_t)NT + cbase + cc;
      float* dst = &tile[r * LDS_STRIDE + cc];
#if USE_ASYNC_LDS
      __builtin_amdgcn_global_load_async_to_lds_b128(
          (gv4i_t*)(void*)src,
          (sv4i_t*)(void*)dst,
          0, 0);
#else
      *(float4*)dst = *(const float4*)src;
#endif
    }
#if USE_ASYNC_LDS
    wait_async_zero();
#endif
    __syncthreads();

    // Each wave consumes its own 16x64 sub-tile: 2 x (16x16x32) WMMA.
    // A-layout (16-bit, 16x32): lanes 0-15 K=0..7,16..23 ; lanes 16-31
    // K=8..15,24..31  -> per-lane 4 x float4 LDS gathers, cvt to f16.
#pragma unroll
    for (int i = 0; i < 2; ++i) {
      const int cb = wave * 64 + i * 32;
      const float* rp = &tile[m * LDS_STRIDE + cb + h * 8];
      const float4 f0 = *(const float4*)(rp + 0);
      const float4 f1 = *(const float4*)(rp + 4);
      const float4 f2 = *(const float4*)(rp + 16);
      const float4 f3 = *(const float4*)(rp + 20);
      v16h a;
      a[0]  = (_Float16)f0.x; a[1]  = (_Float16)f0.y;
      a[2]  = (_Float16)f0.z; a[3]  = (_Float16)f0.w;
      a[4]  = (_Float16)f1.x; a[5]  = (_Float16)f1.y;
      a[6]  = (_Float16)f1.z; a[7]  = (_Float16)f1.w;
      a[8]  = (_Float16)f2.x; a[9]  = (_Float16)f2.y;
      a[10] = (_Float16)f2.z; a[11] = (_Float16)f2.w;
      a[12] = (_Float16)f3.x; a[13] = (_Float16)f3.y;
      a[14] = (_Float16)f3.z; a[15] = (_Float16)f3.w;
      // D = A * ones + C : every column of D holds the row sum.
      acc = __builtin_amdgcn_wmma_f32_16x16x32_f16(
          false, a, false, ones, (short)0, acc, false, false);
    }
    __syncthreads();
  }

  // C/D layout: VGPR j, lane 0 -> (M=j, N=0); lane 16 -> (M=j+8, N=0).
  if (m == 0) {
    const int rbase = (int)row0 + h * 8;
#pragma unroll
    for (int j = 0; j < 8; ++j)
      atomicAdd(&rowsum[rbase + j], acc[j]);
  }
}

// --------------------------- pass 2: envelope -------------------------------
// grid = (NT/(256*4) = 128, NB = 64), block = 256; float4 store per thread.
__global__ __launch_bounds__(256) void adsr_envelope_kernel(
    const float* __restrict__ rowsum,
    const float* __restrict__ attackP,  const float* __restrict__ decayP,
    const float* __restrict__ sustainP, const float* __restrict__ releaseP,
    float* __restrict__ out)
{
  const int b  = blockIdx.y;
  const int t0 = (blockIdx.x * blockDim.x + threadIdx.x) * 4;

  const float attack  = attackP[0];
  const float decay   = decayP[0];
  const float sustain = sustainP[0];
  const float release = releaseP[0];

  const float dtc = expf(-1.0f / decay);     // matches reference decay_tc
  const float rtc = expf(-1.0f / release);   // matches reference release_tc
  const float l2d = log2f(dtc);              // tc^n = exp2(n*log2(tc))
  const float l2r = log2f(rtc);

  const float thrf = rowsum[b];
  const int   thr  = (int)(thrf + 0.5f);     // exact integer count of ones

  // decay_slope at the last gate-on sample (AD = thr), only if it was in
  // the decay region (otherwise the cumsum spike is zero).
  const float ds_last = sustain + (1.0f - sustain) *
                        exp2f(((float)thr - attack) * l2d);
  const float spike = ((float)thr > attack) ? ds_last : 0.0f;
  const float rel0  = sustain + dtc * (spike - sustain);

  float v[4];
#pragma unroll
  for (int k = 0; k < 4; ++k) {
    const int t = t0 + k;
    float val;
    if (t < thr) {                       // gate on: attack or decay
      const float ad = (float)(t + 1);
      val = (ad <= attack)
              ? (ad / attack)
              : sustain + (1.0f - sustain) * exp2f((ad - attack) * l2d);
    } else {                             // gate off: release
      const float r = (float)(t - thr + 1);
      val = rel0 * exp2f(r * l2r);
    }
    v[k] = val;
  }
  float4 o; o.x = v[0]; o.y = v[1]; o.z = v[2]; o.w = v[3];
  *(float4*)(out + (size_t)b * NT + t0) = o;
}

// --------------------------- launcher ---------------------------------------
extern "C" void kernel_launch(void* const* d_in, const int* in_sizes, int n_in,
                              void* d_out, int out_size, void* d_ws, size_t ws_size,
                              hipStream_t stream) {
  (void)in_sizes; (void)n_in; (void)out_size; (void)ws_size;

  const float* gate     = (const float*)d_in[0];
  const float* attackP  = (const float*)d_in[1];
  const float* decayP   = (const float*)d_in[2];
  const float* sustainP = (const float*)d_in[3];
  const float* releaseP = (const float*)d_in[4];
  float* out    = (float*)d_out;
  float* rowsum = (float*)d_ws;   // 64 floats of scratch

  zero_rowsum_kernel<<<1, 64, 0, stream>>>(rowsum);

  dim3 g1(NT / COLS_PER_BLOCK, NB / ROWS);   // (64, 4)
  rowsum_wmma_kernel<<<g1, 256, 0, stream>>>(gate, rowsum);

  dim3 g2(NT / (256 * 4), NB);               // (128, 64)
  adsr_envelope_kernel<<<g2, 256, 0, stream>>>(rowsum, attackP, decayP,
                                               sustainP, releaseP, out);
}
